// ProjectionResBlock_19189913878682
// MI455X (gfx1250) — compile-verified, matrix operand order checked by reading, not run on hardware
//
#include <hip/hip_runtime.h>
#include <hip/hip_bf16.h>

typedef __attribute__((ext_vector_type(16))) _Float16 v16h;
typedef __attribute__((ext_vector_type(8)))  float    v8f;

#define WMMA_F16(a, b, c) \
  __builtin_amdgcn_wmma_f32_16x16x32_f16(false, (a), false, (b), (short)0, (c), false, false)

static constexpr int C_IN  = 128;
static constexpr int C_OUT = 256;
static constexpr int MID   = 64;
static constexpr int KNB   = 27;
static constexpr float EPS = 1e-3f;

// ---------------------------------------------------------------------------
// Elementwise f32 -> f16 convert (row-major copy)
// ---------------------------------------------------------------------------
__global__ void cvt_f16_kernel(const float* __restrict__ x, _Float16* __restrict__ y, size_t n) {
  size_t i = (size_t)blockIdx.x * blockDim.x + threadIdx.x;
  size_t stride = (size_t)gridDim.x * blockDim.x;
  for (; i < n; i += stride) y[i] = (_Float16)x[i];
}

// ---------------------------------------------------------------------------
// Weight transpose + convert: w[b][i][o] (f32) -> wt[b][o][i] (f16)
// ---------------------------------------------------------------------------
__global__ void wcvt_t_kernel(const float* __restrict__ w, _Float16* __restrict__ wt,
                              int B, int I, int O) {
  size_t total = (size_t)B * I * O;
  size_t stride = (size_t)gridDim.x * blockDim.x;
  for (size_t idx = (size_t)blockIdx.x * blockDim.x + threadIdx.x; idx < total; idx += stride) {
    size_t io = (size_t)I * O;
    size_t b = idx / io;
    size_t rem = idx - b * io;
    int i = (int)(rem / O);
    int o = (int)(rem - (size_t)i * O);
    wt[(b * O + o) * I + i] = (_Float16)w[idx];
  }
}

// ---------------------------------------------------------------------------
// Dense GEMM: Y[N, Cout] (f32) = A[N, CIN] (f16) * Bt[Cout, CIN] (f16)
// One wave32 computes a 16x64 output slab: 4 independent accumulators,
// A fragment reused across 4 column tiles (hides WMMA RAW hazard, 4x A reuse).
// ---------------------------------------------------------------------------
template <int CIN>
__global__ __launch_bounds__(256) void gemm_wmma_kernel(
    const _Float16* __restrict__ A, const _Float16* __restrict__ Bt,
    float* __restrict__ Y, int Cout, int totalSlabs) {
  int lane = threadIdx.x & 31;
  int wave = threadIdx.x >> 5;
  int slab = blockIdx.x * (blockDim.x >> 5) + wave;
  if (slab >= totalSlabs) return;
  int slabsPerRow = Cout >> 6;                 // 64-wide slabs
  int rowTile = slab / slabsPerRow;
  int colSlab = slab - rowTile * slabsPerRow;

  int m  = lane & 15;                          // M row (A) / N col (B) within tile
  int kh = (lane >> 4) << 4;                   // K half offset: 0 or 16
  const _Float16* arow  = A  + (size_t)(rowTile * 16 + m) * CIN + kh;
  const _Float16* brow0 = Bt + (size_t)(colSlab * 64 + m) * CIN + kh;

  v8f acc[4] = {};
#pragma unroll
  for (int k0 = 0; k0 < CIN; k0 += 32) {
    v16h a = *(const v16h*)(arow + k0);
#pragma unroll
    for (int t = 0; t < 4; t++) {
      v16h b = *(const v16h*)(brow0 + (size_t)t * 16 * CIN + k0);
      acc[t] = WMMA_F16(a, b, acc[t]);
    }
  }

  int r0 = rowTile * 16 + ((lane >> 4) << 3);
  int c0 = colSlab * 64 + m;
#pragma unroll
  for (int t = 0; t < 4; t++)
#pragma unroll
    for (int i = 0; i < 8; i++)
      Y[(size_t)(r0 + i) * Cout + c0 + t * 16] = acc[t][i];
}

// ---------------------------------------------------------------------------
// Octree 3x3x3 conv: Y[N,64] = sum_k H[neigh[n,k],:] @ W2t[k]  (W2t:[27][64o][64i])
// One wave32 computes 16 nodes x all 64 output channels: each gathered A row
// fragment is reused across 4 column tiles (4x less L2 gather traffic),
// 27*2*4 = 216 WMMAs per wave with 4 independent accumulator chains.
// ---------------------------------------------------------------------------
__global__ __launch_bounds__(256) void octconv_wmma_kernel(
    const _Float16* __restrict__ H, const int* __restrict__ neigh,
    const _Float16* __restrict__ W2t, float* __restrict__ Y, int totalRowTiles) {
  int lane = threadIdx.x & 31;
  int wave = threadIdx.x >> 5;
  int rowTile = blockIdx.x * (blockDim.x >> 5) + wave;
  if (rowTile >= totalRowTiles) return;

  int m  = lane & 15;
  int kh = (lane >> 4) << 4;
  int row = rowTile * 16 + m;                  // node index for this lane's A row

  v8f acc[4] = {};
#pragma unroll
  for (int k = 0; k < KNB; k++) {
    int nb = neigh[(size_t)row * KNB + k];
    const _Float16* hrow = H + (size_t)nb * MID;
    v16h a0 = *(const v16h*)(hrow + kh);
    v16h a1 = *(const v16h*)(hrow + 32 + kh);
    const _Float16* wk = W2t + (size_t)k * MID * MID + (size_t)m * MID + kh;
#pragma unroll
    for (int t = 0; t < 4; t++) {
      v16h b0 = *(const v16h*)(wk + (size_t)t * 16 * MID);
      acc[t] = WMMA_F16(a0, b0, acc[t]);
      v16h b1 = *(const v16h*)(wk + (size_t)t * 16 * MID + 32);
      acc[t] = WMMA_F16(a1, b1, acc[t]);
    }
  }

  int r0 = rowTile * 16 + ((lane >> 4) << 3);
#pragma unroll
  for (int t = 0; t < 4; t++)
#pragma unroll
    for (int i = 0; i < 8; i++)
      Y[(size_t)(r0 + i) * MID + t * 16 + m] = acc[t][i];
}

// ---------------------------------------------------------------------------
// Column stats: per-column sum and sum of squares over N rows (float atomics).
// ---------------------------------------------------------------------------
__global__ void colstats_kernel(const float* __restrict__ Y, int N, int C,
                                float* __restrict__ sum, float* __restrict__ sumsq,
                                int rowsPerBlock) {
  int tpc = blockDim.x / C;                 // threads sharing a column
  int c = threadIdx.x % C;
  int rsub = threadIdx.x / C;
  int r = blockIdx.x * rowsPerBlock + rsub;
  int rend = (blockIdx.x + 1) * rowsPerBlock;
  if (rend > N) rend = N;
  float s = 0.f, ss = 0.f;
  for (; r < rend; r += tpc) {
    float v = Y[(size_t)r * C + c];
    s += v;
    ss += v * v;
  }
  atomicAdd(&sum[c], s);
  atomicAdd(&sumsq[c], ss);
}

__global__ void bnstat_finalize_kernel(const float* __restrict__ sum,
                                       const float* __restrict__ sumsq,
                                       float* __restrict__ mean, float* __restrict__ rstd,
                                       int C, float invN) {
  int c = blockIdx.x * blockDim.x + threadIdx.x;
  if (c < C) {
    float m = sum[c] * invN;
    float v = sumsq[c] * invN - m * m;
    mean[c] = m;
    rstd[c] = rsqrtf(v + EPS);
  }
}

// ---------------------------------------------------------------------------
// BN + ReLU + convert to f16
// ---------------------------------------------------------------------------
__global__ void bn_relu_h16_kernel(const float* __restrict__ Y,
                                   const float* __restrict__ mean, const float* __restrict__ rstd,
                                   const float* __restrict__ g, const float* __restrict__ b,
                                   _Float16* __restrict__ out, size_t total, int C) {
  size_t stride = (size_t)gridDim.x * blockDim.x;
  for (size_t i = (size_t)blockIdx.x * blockDim.x + threadIdx.x; i < total; i += stride) {
    int c = (int)(i % C);
    float v = (Y[i] - mean[c]) * rstd[c] * g[c] + b[c];
    out[i] = (_Float16)fmaxf(v, 0.f);
  }
}

// ---------------------------------------------------------------------------
// Final fuse: out = relu(bn3(y3) + bn_p(yp)); y3 lives in d_out (in place).
// ---------------------------------------------------------------------------
__global__ void fuse_out_kernel(float* __restrict__ Y3, const float* __restrict__ Yp,
                                const float* __restrict__ m3, const float* __restrict__ r3,
                                const float* __restrict__ g3, const float* __restrict__ b3,
                                const float* __restrict__ mp, const float* __restrict__ rp,
                                const float* __restrict__ gp, const float* __restrict__ bp,
                                size_t total) {
  size_t stride = (size_t)gridDim.x * blockDim.x;
  for (size_t i = (size_t)blockIdx.x * blockDim.x + threadIdx.x; i < total; i += stride) {
    int c = (int)(i & (C_OUT - 1));
    float a = (Y3[i] - m3[c]) * r3[c] * g3[c] + b3[c];
    float p = (Yp[i] - mp[c]) * rp[c] * gp[c] + bp[c];
    Y3[i] = fmaxf(a + p, 0.f);
  }
}

// ---------------------------------------------------------------------------
// Host-side orchestration
// ---------------------------------------------------------------------------
extern "C" void kernel_launch(void* const* d_in, const int* in_sizes, int n_in,
                              void* d_out, int out_size, void* d_ws, size_t ws_size,
                              hipStream_t stream) {
  // Inputs (setup_inputs order): x, neigh, depth, w1,g1,b1, w2,g2,b2, w3,g3,b3, wp,gp,bp
  const float* x     = (const float*)d_in[0];
  const int*   neigh = (const int*)d_in[1];
  const float* w1 = (const float*)d_in[3];
  const float* g1 = (const float*)d_in[4];
  const float* b1 = (const float*)d_in[5];
  const float* w2 = (const float*)d_in[6];
  const float* g2 = (const float*)d_in[7];
  const float* b2 = (const float*)d_in[8];
  const float* w3 = (const float*)d_in[9];
  const float* g3 = (const float*)d_in[10];
  const float* b3 = (const float*)d_in[11];
  const float* wp = (const float*)d_in[12];
  const float* gp = (const float*)d_in[13];
  const float* bp = (const float*)d_in[14];

  const size_t N = (size_t)in_sizes[0] / C_IN;   // 200000, divisible by 16
  auto al = [](size_t v) { return (v + 255) & ~(size_t)255; };

  char* ws = (char*)d_ws;
  size_t off = 0;
  _Float16* xh = (_Float16*)(ws + off); off += al(N * C_IN * sizeof(_Float16));
  float*    y  = (float*)(ws + off);    off += al(N * MID * sizeof(float));
  _Float16* hh = (_Float16*)(ws + off); off += al(N * MID * sizeof(_Float16));
  float*    yp = (float*)(ws + off);    off += al(N * C_OUT * sizeof(float));
  _Float16* w1t = (_Float16*)(ws + off); off += al((size_t)C_IN * MID * sizeof(_Float16));
  _Float16* w2t = (_Float16*)(ws + off); off += al((size_t)KNB * MID * MID * sizeof(_Float16));
  _Float16* w3t = (_Float16*)(ws + off); off += al((size_t)MID * C_OUT * sizeof(_Float16));
  _Float16* wpt = (_Float16*)(ws + off); off += al((size_t)C_IN * C_OUT * sizeof(_Float16));
  float* stats = (float*)(ws + off);

  // stats layout (floats)
  float* s1 = stats;        float* ss1 = stats + 64;
  float* s2 = stats + 128;  float* ss2 = stats + 192;
  float* s3 = stats + 256;  float* ss3 = stats + 512;
  float* sp = stats + 768;  float* ssp = stats + 1024;
  float* m1 = stats + 1280; float* r1 = stats + 1344;
  float* m2 = stats + 1408; float* r2 = stats + 1472;
  float* m3 = stats + 1536; float* r3 = stats + 1792;
  float* mp = stats + 2048; float* rp = stats + 2304;

  float* y3 = (float*)d_out;                     // y3 staged in d_out, fused in place
  const float invN = 1.0f / (float)N;

  // Zero the sum/sumsq accumulators (first 1280 floats of stats region).
  hipMemsetAsync(stats, 0, 1280 * sizeof(float), stream);

  // Convert weights (transposed) and activations to f16.
  wcvt_t_kernel<<<64, 256, 0, stream>>>(w1, w1t, 1, C_IN, MID);
  wcvt_t_kernel<<<128, 256, 0, stream>>>(w2, w2t, KNB, MID, MID);
  wcvt_t_kernel<<<64, 256, 0, stream>>>(w3, w3t, 1, MID, C_OUT);
  wcvt_t_kernel<<<64, 256, 0, stream>>>(wp, wpt, 1, C_IN, C_OUT);
  cvt_f16_kernel<<<2048, 256, 0, stream>>>(x, xh, N * C_IN);

  // --- stage 1: y1 = x @ w1, BN+ReLU -> hh (f16) ---
  {
    int slabs = (int)(N / 16) * (MID / 64);      // 64-wide slabs
    gemm_wmma_kernel<C_IN><<<(slabs + 7) / 8, 256, 0, stream>>>(xh, w1t, y, MID, slabs);
    int rpb = 1000, blocks = (int)((N + rpb - 1) / rpb);
    colstats_kernel<<<blocks, 256, 0, stream>>>(y, (int)N, MID, s1, ss1, rpb);
    bnstat_finalize_kernel<<<1, 64, 0, stream>>>(s1, ss1, m1, r1, MID, invN);
    bn_relu_h16_kernel<<<4096, 256, 0, stream>>>(y, m1, r1, g1, b1, hh, N * MID, MID);
  }

  // --- stage 2: gathered 3x3x3 conv, BN+ReLU -> hh (f16, reused) ---
  {
    int rowTiles = (int)(N / 16);
    octconv_wmma_kernel<<<(rowTiles + 7) / 8, 256, 0, stream>>>(hh, neigh, w2t, y, rowTiles);
    int rpb = 1000, blocks = (int)((N + rpb - 1) / rpb);
    colstats_kernel<<<blocks, 256, 0, stream>>>(y, (int)N, MID, s2, ss2, rpb);
    bnstat_finalize_kernel<<<1, 64, 0, stream>>>(s2, ss2, m2, r2, MID, invN);
    bn_relu_h16_kernel<<<4096, 256, 0, stream>>>(y, m2, r2, g2, b2, hh, N * MID, MID);
  }

  // --- stage 3: y3 = h @ w3 (into d_out), yp = x @ wp ---
  {
    int slabs = (int)(N / 16) * (C_OUT / 64);
    gemm_wmma_kernel<MID><<<(slabs + 7) / 8, 256, 0, stream>>>(hh, w3t, y3, C_OUT, slabs);
    gemm_wmma_kernel<C_IN><<<(slabs + 7) / 8, 256, 0, stream>>>(xh, wpt, yp, C_OUT, slabs);
    int rpb = 1000, blocks = (int)((N + rpb - 1) / rpb);
    colstats_kernel<<<blocks, 256, 0, stream>>>(y3, (int)N, C_OUT, s3, ss3, rpb);
    colstats_kernel<<<blocks, 256, 0, stream>>>(yp, (int)N, C_OUT, sp, ssp, rpb);
    bnstat_finalize_kernel<<<1, 256, 0, stream>>>(s3, ss3, m3, r3, C_OUT, invN);
    bnstat_finalize_kernel<<<1, 256, 0, stream>>>(sp, ssp, mp, rp, C_OUT, invN);
  }

  // --- final fuse: out = relu(bn3(y3) + bn_p(yp)) ---
  fuse_out_kernel<<<8192, 256, 0, stream>>>(y3, yp, m3, r3, g3, b3, mp, rp, gp, bp, N * C_OUT);
}
